// DecoderWithPositionLayer_24807731102183
// MI455X (gfx1250) — matrix operand order repaired and checked by reading, not compiled, not run on hardware
//
#include <hip/hip_runtime.h>

typedef _Float16 f16;
typedef __attribute__((ext_vector_type(16))) _Float16 v16h;
typedef __attribute__((ext_vector_type(2)))  _Float16 h2;
typedef __attribute__((ext_vector_type(8)))  float    v8f;
typedef __attribute__((ext_vector_type(4)))  unsigned int v4u;
typedef __attribute__((ext_vector_type(8)))  int v8i;
typedef __attribute__((ext_vector_type(4)))  int v4i;

static constexpr int CB   = 2;     // batch
static constexpr int CSQ  = 256;   // Sq
static constexpr int CSV  = 256;   // Sv (== Sq here)
static constexpr int CD   = 1024;  // model dim
static constexpr int CF   = 64;    // relpos feature dim
static constexpr int CHID = 4096;  // hidden
static constexpr int CHN  = 16;    // heads
static constexpr int CDH  = 64;    // head dim

union AFrag { v16h v; h2 p[8]; };

__device__ __forceinline__ v8f vzero8() {
  v8f z;
  #pragma unroll
  for (int i = 0; i < 8; ++i) z[i] = 0.f;
  return z;
}

// Generic LDS pointer -> 32-bit LDS byte offset (low 32 bits of generic addr).
__device__ __forceinline__ unsigned ldsOff(const void* p) {
  return (unsigned)(size_t)p;
}

// ---- Tensor Data Mover: 2D tile (tile_k x tile_rows) of f16, global->LDS ----
// Builds D# group0/group1 per CDNA5 ISA §8.3/8.4; 2D tensor so groups 2/3 = 0.
// This toolchain uses the 6-arg builtin: (v4u, v8i, v4i, v4i, v8i, i32 cpol).
__device__ __forceinline__ void tdm_load_2d(const f16* gaddr, unsigned lds_addr,
                                            unsigned tile_k, unsigned tile_rows,
                                            unsigned tensor_k, unsigned tensor_rows,
                                            unsigned stride_elems) {
  unsigned long long ga = (unsigned long long)(size_t)gaddr;
  v4u g0;
  g0[0] = 1u;                                   // count=1, user mode, no gather
  g0[1] = lds_addr;                             // lds_addr [63:32]
  g0[2] = (unsigned)(ga & 0xffffffffu);         // global_addr [95:64]
  g0[3] = (unsigned)((ga >> 32) & 0x01ffffffu)  // global_addr [120:96]
        | (2u << 30);                           // type=2 ("image")
  v8i g1;
  g1[0] = (int)(1u << 16);                      // workgroup_mask=0, data_size=1 (2B)
  g1[1] = (int)((tensor_k & 0xffffu) << 16);            // tensor_dim0[15:0]
  g1[2] = (int)(((tensor_k >> 16) & 0xffffu)            // tensor_dim0[31:16]
        | ((tensor_rows & 0xffffu) << 16));             // tensor_dim1[15:0]
  g1[3] = (int)(((tensor_rows >> 16) & 0xffffu)         // tensor_dim1[31:16]
        | ((tile_k & 0xffffu) << 16));                  // tile_dim0
  g1[4] = (int)(tile_rows & 0xffffu);                   // tile_dim1 (tile_dim2=0)
  g1[5] = (int)stride_elems;                            // tensor_dim0_stride[31:0]
  g1[6] = 0;                                            // stride hi, dim1_stride lo
  g1[7] = 0;
  v4i gz4 = {0, 0, 0, 0};
  v8i gz8 = {0, 0, 0, 0, 0, 0, 0, 0};
  __builtin_amdgcn_tensor_load_to_lds(g0, g1, gz4, gz4, gz8, 0);
}

// ---------------- weight convert + transpose: wt[n*K+k] = (f16)w[k*N+n] ----
__global__ void k_cvt_transpose(const float* __restrict__ w, f16* __restrict__ wt,
                                int K, int N) {
  size_t t = (size_t)blockIdx.x * blockDim.x + threadIdx.x;
  if (t >= (size_t)K * N) return;
  int n  = (int)(t / K);
  int kk = (int)(t % K);
  wt[t] = (f16)w[(size_t)kk * N + n];
}

// ---------------- layer norm (f32 in, f16 out), one row per block ----------
__global__ void k_layernorm(const float* __restrict__ x, const float* __restrict__ g,
                            const float* __restrict__ b, f16* __restrict__ y, int D) {
  __shared__ float ssum[8], ssq[8];
  const float* row = x + (size_t)blockIdx.x * D;
  float s = 0.f, sq = 0.f;
  for (int i = threadIdx.x; i < D; i += blockDim.x) {
    float v = row[i]; s += v; sq += v * v;
  }
  #pragma unroll
  for (int off = 16; off > 0; off >>= 1) {
    s  += __shfl_down(s, off);
    sq += __shfl_down(sq, off);
  }
  int wv = threadIdx.x >> 5;
  if ((threadIdx.x & 31) == 0) { ssum[wv] = s; ssq[wv] = sq; }
  __syncthreads();
  if (threadIdx.x == 0) {
    float ts = 0.f, tq = 0.f;
    for (int w2 = 0; w2 < 8; ++w2) { ts += ssum[w2]; tq += ssq[w2]; }
    ssum[0] = ts; ssq[0] = tq;
  }
  __syncthreads();
  float mean = ssum[0] / D;
  float var  = ssq[0] / D - mean * mean;
  float inv  = rsqrtf(var + 1e-3f);
  f16* yrow = y + (size_t)blockIdx.x * D;
  for (int i = threadIdx.x; i < D; i += blockDim.x)
    yrow[i] = (f16)((row[i] - mean) * inv * g[i] + b[i]);
}

// ---------------- TDM double-buffered WMMA GEMM ----------------------------
// C[M,N] = A[M,K](f16) * Bt[N,K](f16)^T + bias.  Block = 8 waves = 128x64 C
// tile; TDM streams 128x32 A and 64x32 B stages into LDS (double buffered),
// wave 0 drives the DMA + TENSORcnt, all waves compute WMMA from LDS.
template<bool RELU, bool OUTF16>
__global__ void k_gemm_tdm(const f16* __restrict__ A, const f16* __restrict__ Bt,
                           const float* __restrict__ bias,
                           float* __restrict__ Cf, f16* __restrict__ Ch,
                           int M, int N, int K) {
  __shared__ f16 As[2][128 * 32];
  __shared__ f16 Bs[2][64 * 32];

  const int lane = threadIdx.x & 31;
  const int lm   = lane & 15;
  const int hf   = lane >> 4;
  const int wv   = threadIdx.x >> 5;

  const int tilesN = N >> 6;
  const int bm = blockIdx.x / tilesN;
  const int bn = blockIdx.x % tilesN;

  const f16* Abase = A  + (size_t)bm * 128 * K;
  const f16* Bbase = Bt + (size_t)bn * 64 * K;

  // prologue: stage 0
  if (wv == 0) {
    tdm_load_2d(Abase, ldsOff(&As[0][0]), 32, 128, (unsigned)K, 128u, (unsigned)K);
    tdm_load_2d(Bbase, ldsOff(&Bs[0][0]), 32, 64,  (unsigned)K, 64u,  (unsigned)K);
    __builtin_amdgcn_s_wait_tensorcnt((short)0);
  }
  __syncthreads();

  v8f acc[4];
  #pragma unroll
  for (int t = 0; t < 4; ++t) acc[t] = vzero8();

  const int nsteps = K >> 5;
  for (int s = 0; s < nsteps; ++s) {
    const int cur = s & 1;
    const int nxt = cur ^ 1;
    // kick off next stage while computing this one
    if (s + 1 < nsteps && wv == 0) {
      tdm_load_2d(Abase + (size_t)(s + 1) * 32, ldsOff(&As[nxt][0]),
                  32, 128, (unsigned)K, 128u, (unsigned)K);
      tdm_load_2d(Bbase + (size_t)(s + 1) * 32, ldsOff(&Bs[nxt][0]),
                  32, 64,  (unsigned)K, 64u,  (unsigned)K);
    }
    // A fragment for this wave's 16-row strip (rows wv*16 .. wv*16+15)
    AFrag a;
    const f16* arow = &As[cur][(wv * 16 + lm) * 32];
    #pragma unroll
    for (int j = 0; j < 8; ++j) {
      int kk = (hf << 4) + (j << 1);
      a.p[j] = *(const h2*)(arow + kk);
    }
    #pragma unroll
    for (int t = 0; t < 4; ++t) {
      AFrag bb;
      const f16* brow = &Bs[cur][(t * 16 + lm) * 32];
      #pragma unroll
      for (int j = 0; j < 8; ++j) {
        int kk = (hf << 4) + (j << 1);
        bb.p[j] = *(const h2*)(brow + kk);
      }
      acc[t] = __builtin_amdgcn_wmma_f32_16x16x32_f16(
          false, a.v, false, bb.v, (short)0, acc[t], false, false);
    }
    if (s + 1 < nsteps) {
      if (wv == 0) __builtin_amdgcn_s_wait_tensorcnt((short)0);
      __syncthreads();   // next stage visible; current buffer free for s+2
    }
  }

  #pragma unroll
  for (int t = 0; t < 4; ++t) {
    int n = bn * 64 + t * 16 + lm;
    float bv = bias[n];
    #pragma unroll
    for (int i = 0; i < 8; ++i) {
      int row = bm * 128 + wv * 16 + hf * 8 + i;
      float v = acc[t][i] + bv;
      if (RELU) v = v > 0.f ? v : 0.f;
      if (OUTF16) Ch[(size_t)row * N + n] = (f16)v;
      else        Cf[(size_t)row * N + n] = v;
    }
  }
}

// ---------------- qkv split: [B,Sq,H*192] -> q/k [B,H,Sq,64], vT [B,H,64,Sq]
__global__ void k_split_qkv(const f16* __restrict__ qkv, f16* __restrict__ q,
                            f16* __restrict__ k, f16* __restrict__ vt) {
  size_t t = (size_t)blockIdx.x * blockDim.x + threadIdx.x;
  size_t total = (size_t)CB * CSQ * CHN * CDH;
  if (t >= total) return;
  int d = (int)(t & 63); size_t r = t >> 6;
  int h = (int)(r & 15); r >>= 4;
  int s = (int)(r % CSQ); int b = (int)(r / CSQ);
  size_t src = ((size_t)(b * CSQ + s)) * (3 * CD) + (size_t)h * 192;
  size_t qi = (((size_t)(b * CHN + h)) * CSQ + s) * CDH + d;
  size_t vi = (((size_t)(b * CHN + h)) * CDH + d) * CSQ + s;
  q[qi]  = qkv[src + d];
  k[qi]  = qkv[src + 64 + d];
  vt[vi] = qkv[src + 128 + d];
}

// ---------------- kv split: [B,Sv,H*128] -> k [B,H,Sv,64], vT [B,H,64,Sv] --
__global__ void k_split_kv(const f16* __restrict__ kv, f16* __restrict__ k,
                           f16* __restrict__ vt) {
  size_t t = (size_t)blockIdx.x * blockDim.x + threadIdx.x;
  size_t total = (size_t)CB * CSV * CHN * CDH;
  if (t >= total) return;
  int d = (int)(t & 63); size_t r = t >> 6;
  int h = (int)(r & 15); r >>= 4;
  int s = (int)(r % CSV); int b = (int)(r / CSV);
  size_t src = ((size_t)(b * CSV + s)) * (2 * CD) + (size_t)h * 128;
  size_t ki = (((size_t)(b * CHN + h)) * CSV + s) * CDH + d;
  size_t vi = (((size_t)(b * CHN + h)) * CDH + d) * CSV + s;
  k[ki]  = kv[src + d];
  vt[vi] = kv[src + 64 + d];
}

// ---------------- [B,Sq,D] -> [B,H,Sq,64] ---------------------------------
__global__ void k_reshape_heads(const f16* __restrict__ y, f16* __restrict__ yf) {
  size_t t = (size_t)blockIdx.x * blockDim.x + threadIdx.x;
  size_t total = (size_t)CB * CSQ * CHN * CDH;
  if (t >= total) return;
  int d = (int)(t & 63); size_t r = t >> 6;
  int h = (int)(r & 15); r >>= 4;
  int s = (int)(r % CSQ); int b = (int)(r / CSQ);
  yf[(((size_t)(b * CHN + h)) * CSQ + s) * CDH + d] =
      y[((size_t)(b * CSQ + s)) * CD + (size_t)h * CDH + d];
}

// ---------------- u = W_h^T q (per b,s,h), c = pos_b_h . q -----------------
__global__ void k_pos_u(const f16* __restrict__ qf, const float* __restrict__ pos_w,
                        const float* __restrict__ pos_b,
                        float* __restrict__ U, float* __restrict__ Cc) {
  int idx = blockIdx.x;                 // (b*Sq + s)*H + h
  int h  = idx % CHN; int bs = idx / CHN;
  int s  = bs % CSQ;  int b  = bs / CSQ;
  const f16* qv = qf + (((size_t)(b * CHN + h)) * CSQ + s) * CDH;
  int f = threadIdx.x;                  // 0..63
  float u = 0.f;
  const float* wrow = pos_w + (size_t)f * CD + (size_t)h * CDH;
  #pragma unroll 8
  for (int d = 0; d < CDH; ++d) u += wrow[d] * (float)qv[d];
  U[(size_t)idx * CF + f] = u;
  if (f == 0) {
    float c = 0.f;
    const float* pb = pos_b + (size_t)h * CDH;
    for (int d = 0; d < CDH; ++d) c += pb[d] * (float)qv[d];
    Cc[idx] = c;
  }
}

// ---------------- bias[b,h,q,k] = relpos[b,q,k,:] . u + c ------------------
__global__ void k_pos_bias(const float* __restrict__ relpos, const float* __restrict__ U,
                           const float* __restrict__ Cc, float* __restrict__ bias) {
  int idx = blockIdx.x;                 // (b*H + h)*Sq + q
  int qq = idx % CSQ; int bh = idx / CSQ;
  int h  = bh % CHN;  int b  = bh / CHN;
  int key = threadIdx.x;                // 0..255
  const float* rp = relpos + (((size_t)(b * CSQ + qq)) * CSQ + key) * CF;
  int uidx = (b * CSQ + qq) * CHN + h;
  const float* u = U + (size_t)uidx * CF;
  float acc = Cc[uidx];
  #pragma unroll 8
  for (int f = 0; f < CF; ++f) acc += rp[f] * u[f];
  bias[(size_t)idx * CSQ + key] = acc;
}

// ---------------- fused attention: one wave per (b,h,16-query tile) --------
// scores via WMMA -> LDS, softmax in-wave, P(f16) -> LDS, PV via WMMA.
template<bool CAUSAL, bool HASBIAS>
__global__ void k_attn(const f16* __restrict__ q, const f16* __restrict__ k,
                       const f16* __restrict__ vt, const float* __restrict__ bias,
                       const unsigned char* __restrict__ kmask,
                       float* __restrict__ out, float scale) {
  constexpr int SK = 256;
  __shared__ float Ssh[2][16][SK];
  __shared__ f16   Psh[2][16][SK];
  __shared__ float Rsh[2][16];

  const int lane = threadIdx.x & 31;
  const int lm = lane & 15, hf = lane >> 4;
  const int wv = threadIdx.x >> 5;
  const int gw = blockIdx.x * 2 + wv;
  const int qt = gw % (CSQ / 16);
  const int bh = gw / (CSQ / 16);
  const int b  = bh / CHN;

  const f16* qb = q  + (((size_t)bh * CSQ) + qt * 16) * CDH;
  const f16* kb = k  + (size_t)bh * SK * CDH;
  const f16* vb = vt + (size_t)bh * CDH * SK;
  const float* biasb = HASBIAS ? bias + (((size_t)bh * CSQ) + qt * 16) * SK : nullptr;
  const unsigned char* km = kmask + (size_t)b * SK;

  // Q fragments (head-dim 64 = two K=32 chunks), reused across all key tiles
  AFrag aq[2];
  const f16* qrow = qb + (size_t)lm * CDH;
  #pragma unroll
  for (int c = 0; c < 2; ++c)
    #pragma unroll
    for (int j = 0; j < 8; ++j) {
      int kk = c * 32 + (hf << 4) + (j << 1);
      aq[c].p[j] = *(const h2*)(qrow + kk);
    }

  // scores S = (Q K^T)*scale + bias, masked -> LDS
  for (int nt = 0; nt < SK / 16; ++nt) {
    v8f acc = vzero8();
    int key = nt * 16 + lm;
    const f16* krow = kb + (size_t)key * CDH;
    #pragma unroll
    for (int c = 0; c < 2; ++c) {
      AFrag bk;
      #pragma unroll
      for (int j = 0; j < 8; ++j) {
        int kk = c * 32 + (hf << 4) + (j << 1);
        bk.p[j] = *(const h2*)(krow + kk);
      }
      acc = __builtin_amdgcn_wmma_f32_16x16x32_f16(
          false, aq[c].v, false, bk.v, (short)0, acc, false, false);
    }
    bool kok = km[key] != 0;
    #pragma unroll
    for (int i = 0; i < 8; ++i) {
      int row = hf * 8 + i;
      float sv = acc[i] * scale;
      if (HASBIAS) sv += biasb[(size_t)row * SK + key];
      if (!kok) sv = -1.0e9f;
      if (CAUSAL && key > qt * 16 + row) sv = -1.0e9f;
      Ssh[wv][row][key] = sv;
    }
  }

  // softmax: lane lm owns row lm; halves split the 256 columns, shfl combine
  {
    float mx = -3.0e38f;
    for (int n = hf * 128; n < hf * 128 + 128; ++n) mx = fmaxf(mx, Ssh[wv][lm][n]);
    mx = fmaxf(mx, __shfl_xor(mx, 16));
    float sum = 0.f;
    for (int n = hf * 128; n < hf * 128 + 128; ++n) {
      float p = __expf(Ssh[wv][lm][n] - mx);
      Psh[wv][lm][n] = (f16)p;
      sum += p;
    }
    sum += __shfl_xor(sum, 16);
    if (hf == 0) Rsh[wv][lm] = sum;
  }

  // out = P @ V (V pre-transposed [64, SK] so B-frags load contiguously)
  v8f ao[4];
  #pragma unroll
  for (int t = 0; t < 4; ++t) ao[t] = vzero8();
  for (int kc = 0; kc < SK / 32; ++kc) {
    AFrag ap;
    #pragma unroll
    for (int j = 0; j < 8; ++j) {
      int kk = kc * 32 + (hf << 4) + (j << 1);
      ap.p[j] = *(const h2*)(&Psh[wv][lm][kk]);
    }
    #pragma unroll
    for (int t = 0; t < 4; ++t) {
      AFrag bv;
      const f16* vrow = vb + (size_t)(t * 16 + lm) * SK;
      #pragma unroll
      for (int j = 0; j < 8; ++j) {
        int kk = kc * 32 + (hf << 4) + (j << 1);
        bv.p[j] = *(const h2*)(vrow + kk);
      }
      ao[t] = __builtin_amdgcn_wmma_f32_16x16x32_f16(
          false, ap.v, false, bv.v, (short)0, ao[t], false, false);
    }
  }

  float* ob = out + (((size_t)bh * CSQ) + qt * 16) * CDH;
  #pragma unroll
  for (int t = 0; t < 4; ++t) {
    int d = t * 16 + lm;
    #pragma unroll
    for (int i = 0; i < 8; ++i) {
      int row = hf * 8 + i;
      ob[(size_t)row * CDH + d] = ao[t][i] / Rsh[wv][row];
    }
  }
}

// ---------------- residual add with head-permute ---------------------------
__global__ void k_add_attn(const float* __restrict__ base, const float* __restrict__ att,
                           float* __restrict__ out) {
  size_t t = (size_t)blockIdx.x * blockDim.x + threadIdx.x;
  size_t total = (size_t)CB * CSQ * CHN * CDH;
  if (t >= total) return;
  int d = (int)(t & 63); size_t r = t >> 6;
  int h = (int)(r & 15); r >>= 4;
  int s = (int)(r % CSQ); int b = (int)(r / CSQ);
  size_t oi = ((size_t)(b * CSQ + s)) * CD + (size_t)h * CDH + d;
  size_t ai = (((size_t)(b * CHN + h)) * CSQ + s) * CDH + d;
  out[oi] = base[oi] + att[ai];
}

__global__ void k_add(const float* __restrict__ a, const float* __restrict__ b,
                      float* __restrict__ o, size_t n) {
  size_t t = (size_t)blockIdx.x * blockDim.x + threadIdx.x;
  if (t < n) o[t] = a[t] + b[t];
}

extern "C" void kernel_launch(void* const* d_in, const int* in_sizes, int n_in,
                              void* d_out, int out_size, void* d_ws, size_t ws_size,
                              hipStream_t stream) {
  (void)in_sizes; (void)n_in; (void)out_size; (void)ws_size;
  auto P = [&](int i) { return (const float*)d_in[i]; };
  const float* queries = P(0);
  const float* values  = P(1);
  const unsigned char* qmask = (const unsigned char*)d_in[2];
  const unsigned char* vmask = (const unsigned char*)d_in[3];
  const float* relpos  = P(4);
  const float* pos_w   = P(29);
  const float* pos_b   = P(30);

  // bump allocator over workspace (deterministic, no hipMalloc)
  char* wsp = (char*)d_ws;
  auto alloc = [&](size_t bytes) -> void* {
    void* r = (void*)wsp;
    wsp += (bytes + 255) & ~(size_t)255;
    return r;
  };

  const int M = CB * CSQ;  // 512 rows
  const int w2n[4] = {3 * CD, CD, 2 * CD, CD};
  f16* w1t[4]; f16* w2t[4];
  for (int i = 0; i < 4; ++i) {
    w1t[i] = (f16*)alloc((size_t)CD * CHID * 2);
    w2t[i] = (f16*)alloc((size_t)CHID * w2n[i] * 2);
  }
  f16* lnb  = (f16*)alloc((size_t)M * CD * 2);
  f16* hb   = (f16*)alloc((size_t)M * CHID * 2);
  f16* qkvb = (f16*)alloc((size_t)M * 3 * CD * 2);
  f16* kvb  = (f16*)alloc((size_t)M * 2 * CD * 2);
  f16* yb   = (f16*)alloc((size_t)M * CD * 2);
  f16* qf   = (f16*)alloc((size_t)M * CD * 2);
  f16* kf   = (f16*)alloc((size_t)M * CD * 2);
  f16* vtf  = (f16*)alloc((size_t)M * CD * 2);
  f16* yf   = (f16*)alloc((size_t)M * CD * 2);
  f16* kcf  = (f16*)alloc((size_t)M * CD * 2);
  f16* vctf = (f16*)alloc((size_t)M * CD * 2);
  float* Ub    = (float*)alloc((size_t)CB * CSQ * CHN * CF * 4);
  float* Ccb   = (float*)alloc((size_t)CB * CSQ * CHN * 4);
  float* biasb = (float*)alloc((size_t)CB * CHN * CSQ * CSQ * 4);
  float* aout  = (float*)alloc((size_t)M * CD * 4);
  float* cur   = (float*)alloc((size_t)M * CD * 4);
  float* o3    = (float*)alloc((size_t)M * CD * 4);

  // --- convert all 8 MLP weight matrices to transposed f16 ---
  const float* w1s[4] = {P(7), P(13), P(19), P(25)};
  const float* w2s[4] = {P(9), P(15), P(21), P(27)};
  for (int i = 0; i < 4; ++i) {
    size_t t1 = (size_t)CD * CHID;
    k_cvt_transpose<<<(unsigned)((t1 + 255) / 256), 256, 0, stream>>>(w1s[i], w1t[i], CD, CHID);
    size_t t2 = (size_t)CHID * w2n[i];
    k_cvt_transpose<<<(unsigned)((t2 + 255) / 256), 256, 0, stream>>>(w2s[i], w2t[i], CHID, w2n[i]);
  }

  auto gemm = [&](const f16* A, const f16* Bt, const float* bias, float* Cf, f16* Ch,
                  int Mm, int N, int K, bool relu, bool outf16) {
    int blocks = (Mm / 128) * (N / 64);
    if (relu)
      k_gemm_tdm<true, true><<<blocks, 256, 0, stream>>>(A, Bt, bias, nullptr, Ch, Mm, N, K);
    else if (outf16)
      k_gemm_tdm<false, true><<<blocks, 256, 0, stream>>>(A, Bt, bias, nullptr, Ch, Mm, N, K);
    else
      k_gemm_tdm<false, false><<<blocks, 256, 0, stream>>>(A, Bt, bias, Cf, nullptr, Mm, N, K);
  };

  const size_t tot = (size_t)M * CD;
  const unsigned eb = (unsigned)((tot + 255) / 256);

  // ---- block0: qkv projection ----
  k_layernorm<<<M, 256, 0, stream>>>(queries, P(5), P(6), lnb, CD);
  gemm(lnb, w1t[0], P(8), nullptr, hb, M, CHID, CD, true, true);
  gemm(hb, w2t[0], P(10), nullptr, qkvb, M, 3 * CD, CHID, false, true);
  k_split_qkv<<<eb, 256, 0, stream>>>(qkvb, qf, kf, vtf);

  // ---- factored position bias: bias = relpos.(W_h q) + pos_b_h.q ----
  k_pos_u<<<CB * CSQ * CHN, CF, 0, stream>>>(qf, pos_w, pos_b, Ub, Ccb);
  k_pos_bias<<<CB * CHN * CSQ, CSQ, 0, stream>>>(relpos, Ub, Ccb, biasb);

  // ---- causal self-attention with bias ----
  k_attn<true, true><<<CB * CHN * (CSQ / 16) / 2, 64, 0, stream>>>(
      qf, kf, vtf, biasb, qmask, aout, 0.125f);
  k_add_attn<<<eb, 256, 0, stream>>>(queries, aout, cur);

  // ---- block1: query transform for cross-attn ----
  k_layernorm<<<M, 256, 0, stream>>>(cur, P(11), P(12), lnb, CD);
  gemm(lnb, w1t[1], P(14), nullptr, hb, M, CHID, CD, true, true);
  gemm(hb, w2t[1], P(16), nullptr, yb, M, CD, CHID, false, true);
  k_reshape_heads<<<eb, 256, 0, stream>>>(yb, yf);

  // ---- block2: kv from values ----
  k_layernorm<<<M, 256, 0, stream>>>(values, P(17), P(18), lnb, CD);
  gemm(lnb, w1t[2], P(20), nullptr, hb, M, CHID, CD, true, true);
  gemm(hb, w2t[2], P(22), nullptr, kvb, M, 2 * CD, CHID, false, true);
  k_split_kv<<<eb, 256, 0, stream>>>(kvb, kcf, vctf);

  // ---- cross-attention ----
  k_attn<false, false><<<CB * CHN * (CSQ / 16) / 2, 64, 0, stream>>>(
      yf, kcf, vctf, nullptr, vmask, aout, 0.125f);
  k_add_attn<<<eb, 256, 0, stream>>>(cur, aout, cur);

  // ---- block3: final MLP + residual ----
  k_layernorm<<<M, 256, 0, stream>>>(cur, P(23), P(24), lnb, CD);
  gemm(lnb, w1t[3], P(26), nullptr, hb, M, CHID, CD, true, true);
  gemm(hb, w2t[3], P(28), o3, nullptr, M, CD, CHID, false, false);
  k_add<<<eb, 256, 0, stream>>>(cur, o3, (float*)d_out, tot);
}
